// TransformerLayer_5763846111742
// MI455X (gfx1250) — matrix operand order repaired and checked by reading, compiled, not run on hardware
//
#include <hip/hip_runtime.h>
#include <stdint.h>

// ---------------------------------------------------------------------------
// CDNA5 (gfx1250) transformer layer: all GEMMs + attention via
// v_wmma_f32_16x16x32_bf16. wave32 layouts per CDNA5 ISA.
// ---------------------------------------------------------------------------

typedef __attribute__((ext_vector_type(16))) __bf16 v16bf;
typedef __attribute__((ext_vector_type(8)))  float  v8f;

#define WMMA_BF16(a, b, c) \
  __builtin_amdgcn_wmma_f32_16x16x32_bf16(false, (a), false, (b), (short)0, (c), false, false)

static __device__ __forceinline__ v8f v8f_zero() {
  v8f z = {0.f, 0.f, 0.f, 0.f, 0.f, 0.f, 0.f, 0.f};
  return z;
}

// round-to-nearest-even f32 -> bf16
static __device__ __forceinline__ unsigned short f2bf(float f) {
  unsigned int u = __float_as_uint(f);
  u += 0x7FFFu + ((u >> 16) & 1u);
  return (unsigned short)(u >> 16);
}

union Frag {
  v16bf v;
  uint4 q[2];
};

// A-fragment 16x32 (MxK) bf16. Lane L: row m = L%16, h = L/16.
// elems 0..15 -> K = h*16 + i  (32 contiguous bytes)
// Caller passes p0 = &row[kbase + h*16].
static __device__ __forceinline__ v16bf load_a_frag(const unsigned short* p0) {
  Frag f;
  f.q[0] = *(const uint4*)(p0);
  f.q[1] = *(const uint4*)(p0 + 8);
  return f.v;
}

// B-fragment 32x16 (KxN) bf16. Lane L: col n = L%16, h = L/16.
// elems 0..15 -> K = h*16 + i  (32 contiguous bytes)
// Caller passes p0 = &col_row[kbase + h*16] (K-major storage, i.e. W[n][k]).
static __device__ __forceinline__ v16bf load_b_frag(const unsigned short* p0) {
  Frag f;
  f.q[0] = *(const uint4*)(p0);
  f.q[1] = *(const uint4*)(p0 + 8);
  return f.v;
}

// ---------------------------------------------------------------------------
// fp32 -> bf16 conversion (weights)
// ---------------------------------------------------------------------------
__global__ __launch_bounds__(256) void tl_f32_to_bf16(const float* __restrict__ s,
                                                      unsigned short* __restrict__ d, int n) {
  int i = blockIdx.x * 256 + threadIdx.x;
  if (i < n) d[i] = f2bf(s[i]);
}

// ---------------------------------------------------------------------------
// LayerNorm over E=1024, one row per block, bf16 output
// ---------------------------------------------------------------------------
__global__ __launch_bounds__(256) void tl_layernorm_bf16(const float* __restrict__ x,
                                                         const float* __restrict__ g,
                                                         const float* __restrict__ b,
                                                         unsigned short* __restrict__ y) {
  __shared__ float rs[256], rs2[256];
  const int row = blockIdx.x, tid = threadIdx.x;
  const float* xr = x + (size_t)row * 1024;
  float v0 = xr[tid], v1 = xr[tid + 256], v2 = xr[tid + 512], v3 = xr[tid + 768];
  rs[tid]  = v0 + v1 + v2 + v3;
  rs2[tid] = v0 * v0 + v1 * v1 + v2 * v2 + v3 * v3;
  __syncthreads();
  for (int off = 128; off > 0; off >>= 1) {
    if (tid < off) { rs[tid] += rs[tid + off]; rs2[tid] += rs2[tid + off]; }
    __syncthreads();
  }
  const float mean = rs[0] * (1.f / 1024.f);
  const float var  = rs2[0] * (1.f / 1024.f) - mean * mean;
  const float r    = rsqrtf(var + 1e-5f);
  unsigned short* yr = y + (size_t)row * 1024;
  yr[tid]       = f2bf((v0 - mean) * r * g[tid]       + b[tid]);
  yr[tid + 256] = f2bf((v1 - mean) * r * g[tid + 256] + b[tid + 256]);
  yr[tid + 512] = f2bf((v2 - mean) * r * g[tid + 512] + b[tid + 512]);
  yr[tid + 768] = f2bf((v3 - mean) * r * g[tid + 768] + b[tid + 768]);
}

// ---------------------------------------------------------------------------
// NT GEMM: C[m,n] = sum_k A[m,k]*W[n,k] + bias[n]  (A, W bf16 K-major)
// MODE 0: bf16 out; MODE 1: relu -> bf16 out; MODE 2: f32 out + resid
// block = 128 thr (4 waves, 2x2), block tile 64(M) x 128(N), wave tile 32x64
// One-stage software pipeline: next K-tile's fragments are in flight while
// the current tile's 8 WMMAs issue.
// ---------------------------------------------------------------------------
template <int MODE>
__global__ __launch_bounds__(128) void tl_gemm_nt(const unsigned short* __restrict__ A,
                                                  const unsigned short* __restrict__ W,
                                                  const float* __restrict__ bias,
                                                  const float* __restrict__ resid,
                                                  void* __restrict__ out,
                                                  int M, int N, int K) {
  const int lane = threadIdx.x & 31;
  const int wave = threadIdx.x >> 5;
  const int n0 = lane & 15, hh = lane >> 4;
  const int mb = blockIdx.y * 64 + (wave & 1) * 32;
  const int nb = blockIdx.x * 128 + (wave >> 1) * 64;

  v8f acc[2][4];
#pragma unroll
  for (int mi = 0; mi < 2; ++mi)
#pragma unroll
    for (int ni = 0; ni < 4; ++ni) acc[mi][ni] = v8f_zero();

  const unsigned short* aptr0 = A + (size_t)(mb + n0) * K + hh * 16;
  const unsigned short* aptr1 = aptr0 + (size_t)16 * K;
  const unsigned short* bptr0 = W + (size_t)(nb + n0) * K + hh * 16;
  const unsigned short* bptr1 = bptr0 + (size_t)16 * K;
  const unsigned short* bptr2 = bptr0 + (size_t)32 * K;
  const unsigned short* bptr3 = bptr0 + (size_t)48 * K;

  v16bf a0 = load_a_frag(aptr0);
  v16bf a1 = load_a_frag(aptr1);
  v16bf b0 = load_b_frag(bptr0);
  v16bf b1 = load_b_frag(bptr1);
  v16bf b2 = load_b_frag(bptr2);
  v16bf b3 = load_b_frag(bptr3);

  for (int k0 = 0;; k0 += 32) {
    const int kn = (k0 + 32 < K) ? (k0 + 32) : k0;   // last iter reloads (dead)
    v16bf na0 = load_a_frag(aptr0 + kn);
    v16bf na1 = load_a_frag(aptr1 + kn);
    v16bf nb0 = load_b_frag(bptr0 + kn);
    v16bf nb1 = load_b_frag(bptr1 + kn);
    v16bf nb2 = load_b_frag(bptr2 + kn);
    v16bf nb3 = load_b_frag(bptr3 + kn);

    acc[0][0] = WMMA_BF16(a0, b0, acc[0][0]);
    acc[1][0] = WMMA_BF16(a1, b0, acc[1][0]);
    acc[0][1] = WMMA_BF16(a0, b1, acc[0][1]);
    acc[1][1] = WMMA_BF16(a1, b1, acc[1][1]);
    acc[0][2] = WMMA_BF16(a0, b2, acc[0][2]);
    acc[1][2] = WMMA_BF16(a1, b2, acc[1][2]);
    acc[0][3] = WMMA_BF16(a0, b3, acc[0][3]);
    acc[1][3] = WMMA_BF16(a1, b3, acc[1][3]);

    if (k0 + 32 >= K) break;
    a0 = na0; a1 = na1; b0 = nb0; b1 = nb1; b2 = nb2; b3 = nb3;
  }

  float* fout = (float*)out;
  unsigned short* bout = (unsigned short*)out;
#pragma unroll
  for (int mi = 0; mi < 2; ++mi) {
#pragma unroll
    for (int ni = 0; ni < 4; ++ni) {
      const int col = nb + ni * 16 + n0;
      const float bv = bias[col];
#pragma unroll
      for (int i = 0; i < 8; ++i) {
        const int row = mb + mi * 16 + hh * 8 + i;   // C layout: m = h*8 + i
        float v = acc[mi][ni][i] + bv;
        if (MODE == 1) v = fmaxf(v, 0.f);
        const size_t idx = (size_t)row * N + col;
        if (MODE == 2) fout[idx] = v + resid[idx];
        else           bout[idx] = f2bf(v);
      }
    }
  }
}

// ---------------------------------------------------------------------------
// V transpose: qkv[b,s, 2048 + h*64 + d] -> vT[(b*16+h)*64 + d, s]
// ---------------------------------------------------------------------------
__global__ __launch_bounds__(256) void tl_transpose_v(const unsigned short* __restrict__ qkv,
                                                      unsigned short* __restrict__ vT) {
  const int t = blockIdx.x * 256 + threadIdx.x;   // 4*16*64*1024 threads
  const int s = t & 1023;
  const int d = (t >> 10) & 63;
  const int bh = t >> 16;
  const int head = bh & 15, batch = bh >> 4;
  const unsigned short v = qkv[((size_t)(batch * 1024 + s)) * 3072 + 2048 + head * 64 + d];
  vT[((size_t)bh * 64 + d) * 1024 + s] = v;
}

// ---------------------------------------------------------------------------
// One 32-key block of causal flash attention.
// MASKED=false: interior block, all keys valid (no causal test at all).
// MASKED=true : diagonal block, branchless clamp + select (no EXEC divergence).
// ---------------------------------------------------------------------------
template <bool MASKED>
static __device__ __forceinline__ void attn_kb_block(
    int kb, int qb, int n0, int hh,
    const unsigned short* __restrict__ ksec, const unsigned short* __restrict__ vtb,
    const float* __restrict__ rp, v16bf qa0, v16bf qa1, v16bf ones,
    unsigned short* ldsRow, float* m_i, v8f* accO, v8f& accS) {
  // ---- scores S = Q K^T (two 16-key groups) ----
  v8f s0 = v8f_zero(), s1 = v8f_zero();
  {
    const unsigned short* kp0 = ksec + (size_t)(kb * 32 + n0) * 3072;
    const unsigned short* kp1 = kp0 + (size_t)16 * 3072;
    s0 = WMMA_BF16(qa0, load_b_frag(kp0 + hh * 16), s0);
    s0 = WMMA_BF16(qa1, load_b_frag(kp0 + 32 + hh * 16), s0);
    s1 = WMMA_BF16(qa0, load_b_frag(kp1 + hh * 16), s1);
    s1 = WMMA_BF16(qa1, load_b_frag(kp1 + 32 + hh * 16), s1);
  }
  // ---- rel-pos bias: lane's 8 indices are consecutive (q-j = qi0+i-j) ----
  const int j0 = kb * 32 + n0;
  const int qi0 = qb * 16 + hh * 8;
  float bias0[8], bias1[8];
  if (!MASKED) {
    const float* rp0 = rp + (qi0 - j0);
#pragma unroll
    for (int i = 0; i < 8; ++i) { bias0[i] = rp0[i]; bias1[i] = rp0[i - 16]; }
  } else {
#pragma unroll
    for (int i = 0; i < 8; ++i) {
      const int d0 = qi0 + i - j0;
      const int d1 = d0 - 16;
      const float r0 = rp[d0 >= 0 ? d0 : 0];
      const float r1 = rp[d1 >= 0 ? d1 : 0];
      bias0[i] = (d0 >= 0) ? r0 : -3.0e38f;
      bias1[i] = (d1 >= 0) ? r1 : -3.0e38f;
    }
  }
  // ---- scale + bias, running row max ----
  float sv0[8], sv1[8], nm[8];
#pragma unroll
  for (int i = 0; i < 8; ++i) {
    const float a = s0[i] * 0.125f + bias0[i];
    const float c = s1[i] * 0.125f + bias1[i];
    sv0[i] = a; sv1[i] = c;
    nm[i] = fmaxf(m_i[i], fmaxf(a, c));
  }
#pragma unroll
  for (int msk = 1; msk < 16; msk <<= 1)
#pragma unroll
    for (int i = 0; i < 8; ++i) nm[i] = fmaxf(nm[i], __shfl_xor(nm[i], msk, 32));
  // ---- exp, stash P to LDS (C layout -> row-major 16x32) ----
#pragma unroll
  for (int i = 0; i < 8; ++i) {
    ldsRow[(hh * 8 + i) * 32 + n0]      = f2bf(__expf(sv0[i] - nm[i]));
    ldsRow[(hh * 8 + i) * 32 + 16 + n0] = f2bf(__expf(sv1[i] - nm[i]));
  }
  // ---- rescale running O and row-sum accumulators ----
#pragma unroll
  for (int i = 0; i < 8; ++i) {
    const float f = __expf(m_i[i] - nm[i]);
    m_i[i] = nm[i];
    accS[i] = accS[i] * f;
#pragma unroll
    for (int ni = 0; ni < 4; ++ni) accO[ni][i] = accO[ni][i] * f;
  }
  // same-wave LDS ops are in-order; wait defensively before the re-read
  asm volatile("s_wait_dscnt 0x0" ::: "memory");
  // ---- A = P (16x32) from LDS; row sums via P*ones; O += P*V ----
  const v16bf pa = load_a_frag(&ldsRow[n0 * 32 + hh * 16]);
  accS = WMMA_BF16(pa, ones, accS);
#pragma unroll
  for (int ni = 0; ni < 4; ++ni) {
    const unsigned short* vp = vtb + (size_t)(ni * 16 + n0) * 1024 + kb * 32 + hh * 16;
    accO[ni] = WMMA_BF16(pa, load_b_frag(vp), accO[ni]);
  }
}

// ---------------------------------------------------------------------------
// Causal flash attention with relative-position bias.
// One wave per (b, h, 16-query block); key blocks of 32; interior blocks run
// mask-free, only the final diagonal block applies the causal select.
// ---------------------------------------------------------------------------
__global__ __launch_bounds__(256) void tl_attention(const unsigned short* __restrict__ qkv,
                                                    const unsigned short* __restrict__ vT,
                                                    const float* __restrict__ rel_pos,
                                                    unsigned short* __restrict__ o) {
  __shared__ unsigned short ldsP[8][16 * 32];     // per-wave 16x32 P tile (bf16)
  const int lane = threadIdx.x & 31;
  const int wave = threadIdx.x >> 5;
  const int n0 = lane & 15, hh = lane >> 4;

  const int task = blockIdx.x * 8 + wave;         // 4096 tasks total
  const int qb = task & 63;                       // query block (S/16 = 64)
  const int bh = task >> 6;
  const int head = bh & 15, batch = bh >> 4;

  const unsigned short* qsec = qkv + (size_t)batch * 1024 * 3072 + head * 64;
  const unsigned short* ksec = qsec + 1024;
  const unsigned short* vtb  = vT + (size_t)bh * 64 * 1024;
  const float* rp = rel_pos + head * 2048;

  // all-ones B fragment (bf16 1.0 = 0x3F80): accS += P * ones => row sums
  Frag onesf;
  onesf.q[0] = make_uint4(0x3F803F80u, 0x3F803F80u, 0x3F803F80u, 0x3F803F80u);
  onesf.q[1] = onesf.q[0];
  const v16bf ones = onesf.v;

  // Q A-fragments for the whole loop (rows = 16 queries, K = head dim 0..63)
  const unsigned short* qp = qsec + (size_t)(qb * 16 + n0) * 3072;
  const v16bf qa0 = load_a_frag(qp + hh * 16);           // d 0..31
  const v16bf qa1 = load_a_frag(qp + 32 + hh * 16);      // d 32..63

  float m_i[8];
#pragma unroll
  for (int i = 0; i < 8; ++i) m_i[i] = -3.0e38f;
  v8f accO[4];
#pragma unroll
  for (int ni = 0; ni < 4; ++ni) accO[ni] = v8f_zero();
  v8f accS = v8f_zero();                          // running row sums (softmax denom)

  // interior key blocks: max key = 32kb+31 <= 16qb-1 < q  -> never masked
  const int kb_diag = qb >> 1;
  for (int kb = 0; kb < kb_diag; ++kb)
    attn_kb_block<false>(kb, qb, n0, hh, ksec, vtb, rp, qa0, qa1, ones,
                         &ldsP[wave][0], m_i, accO, accS);
  // diagonal block: branchless causal mask
  attn_kb_block<true>(kb_diag, qb, n0, hh, ksec, vtb, rp, qa0, qa1, ones,
                      &ldsP[wave][0], m_i, accO, accS);

  // ---- normalize and write O (token-major, E contiguous) ----
  // accS columns all hold the row sum, so each lane already has its rows' denom.
#pragma unroll
  for (int i = 0; i < 8; ++i) {
    const int q = qb * 16 + hh * 8 + i;
    const float inv = 1.0f / accS[i];
    const size_t base = (size_t)(batch * 1024 + q) * 1024 + head * 64;
#pragma unroll
    for (int ni = 0; ni < 4; ++ni)
      o[base + ni * 16 + n0] = f2bf(accO[ni][i] * inv);
  }
}

// ---------------------------------------------------------------------------
// host-side orchestration
// ---------------------------------------------------------------------------
extern "C" void kernel_launch(void* const* d_in, const int* in_sizes, int n_in,
                              void* d_out, int out_size, void* d_ws, size_t ws_size,
                              hipStream_t stream) {
  (void)in_sizes; (void)n_in; (void)out_size; (void)ws_size;
  const float* x       = (const float*)d_in[0];
  const float* rel_pos = (const float*)d_in[1];
  const float* in_w    = (const float*)d_in[2];
  const float* in_b    = (const float*)d_in[3];
  const float* out_w   = (const float*)d_in[4];
  const float* out_b   = (const float*)d_in[5];
  const float* w1      = (const float*)d_in[6];
  const float* b1      = (const float*)d_in[7];
  const float* w2      = (const float*)d_in[8];
  const float* b2      = (const float*)d_in[9];
  const float* ln1_g   = (const float*)d_in[10];
  const float* ln1_b   = (const float*)d_in[11];
  const float* ln2_g   = (const float*)d_in[12];
  const float* ln2_b   = (const float*)d_in[13];
  float* out = (float*)d_out;

  char* ws = (char*)d_ws;
  const size_t MB = 1024 * 1024;
  // layout (lifetimes noted); total <= 88 MB
  unsigned short* wq   = (unsigned short*)(ws +  0 * MB); // 6 MB  in_proj_w bf16
  unsigned short* wo   = (unsigned short*)(ws +  6 * MB); // 2 MB  out_w bf16
  unsigned short* w1b  = (unsigned short*)(ws +  8 * MB); // 8 MB  w1 bf16
  unsigned short* w2b  = (unsigned short*)(ws + 16 * MB); // 8 MB  w2 bf16
  unsigned short* qkvb = (unsigned short*)(ws + 24 * MB); // 24 MB (dead after attention)
  unsigned short* vTp  = (unsigned short*)(ws + 48 * MB); // 8 MB  (dead after attention)
  unsigned short* ob   = (unsigned short*)(ws + 56 * MB); // 8 MB  attention out
  float*          x2   = (float*)        (ws + 64 * MB); // 16 MB x + attn (residual)
  unsigned short* xn   = (unsigned short*)(ws + 80 * MB); // 8 MB  ln1 out / ln2 out (reused)
  unsigned short* h1   = (unsigned short*)(ws + 24 * MB); // 32 MB MLP hidden, aliases qkvb+vT

  // 1) weights -> bf16
  tl_f32_to_bf16<<<(3072 * 1024) / 256, 256, 0, stream>>>(in_w, wq, 3072 * 1024);
  tl_f32_to_bf16<<<(1024 * 1024) / 256, 256, 0, stream>>>(out_w, wo, 1024 * 1024);
  tl_f32_to_bf16<<<(4096 * 1024) / 256, 256, 0, stream>>>(w1, w1b, 4096 * 1024);
  tl_f32_to_bf16<<<(4096 * 1024) / 256, 256, 0, stream>>>(w2, w2b, 4096 * 1024);

  // 2) ln1 -> bf16
  tl_layernorm_bf16<<<4096, 256, 0, stream>>>(x, ln1_g, ln1_b, xn);

  // 3) qkv = ln1(x) @ in_proj_w.T + in_proj_b   [4096 x 3072] bf16
  tl_gemm_nt<0><<<dim3(3072 / 128, 4096 / 64), 128, 0, stream>>>(
      xn, wq, in_b, nullptr, qkvb, 4096, 3072, 1024);

  // 4) V transpose for contiguous P*V B-fragments
  tl_transpose_v<<<(4 * 16 * 64 * 1024) / 256, 256, 0, stream>>>(qkvb, vTp);

  // 5) causal attention with rel-pos bias -> ob [4096 x 1024] bf16
  tl_attention<<<512, 256, 0, stream>>>(qkvb, vTp, rel_pos, ob);

  // 6) x2 = x + ob @ out_w.T + out_b   (f32)
  tl_gemm_nt<2><<<dim3(1024 / 128, 4096 / 64), 128, 0, stream>>>(
      ob, wo, out_b, x, x2, 4096, 1024, 1024);

  // 7) ln2 -> bf16 (reuse xn buffer)
  tl_layernorm_bf16<<<4096, 256, 0, stream>>>(x2, ln2_g, ln2_b, xn);

  // 8) h1 = relu(ln2 @ w1.T + b1)   [4096 x 4096] bf16
  tl_gemm_nt<1><<<dim3(4096 / 128, 4096 / 64), 128, 0, stream>>>(
      xn, w1b, b1, nullptr, h1, 4096, 4096, 1024);

  // 9) out = x2 + h1 @ w2.T + b2   (f32, K = 4096)
  tl_gemm_nt<2><<<dim3(1024 / 128, 4096 / 64), 128, 0, stream>>>(
      h1, w2b, b2, x2, out, 4096, 1024, 4096);
}
